// _LPNFBase_71433896067564
// MI455X (gfx1250) — matrix operand order repaired and checked by reading, not compiled, not run on hardware
//
#include <hip/hip_runtime.h>
#include <hip/hip_bf16.h>
#include <stdint.h>

// H = hidden = 128, K = 64 nodes/subgraph, 256 threads = 8 wave32 per block.
#define HH 128
#define KK 64
#define TPB 256

typedef __attribute__((ext_vector_type(16))) __bf16 v16bf;
typedef __attribute__((ext_vector_type(8)))  float  f32x8;

union Frag {
    uint32_t u[8];
    uint4    q[2];
    v16bf    v;
};

__device__ __forceinline__ unsigned short f2bf(float f) {
    union { float f; uint32_t u; } x; x.f = f;
    uint32_t r = x.u + 0x7FFFu + ((x.u >> 16) & 1u);   // round-nearest-even
    return (unsigned short)(r >> 16);
}
__device__ __forceinline__ float bf2f(unsigned short h) {
    union { uint32_t u; float f; } x; x.u = ((uint32_t)h) << 16; return x.f;
}
__device__ __forceinline__ f32x8 zero8() {
    f32x8 z;
#pragma unroll
    for (int i = 0; i < 8; ++i) z[i] = 0.f;
    return z;
}
__device__ __forceinline__ void atomAddF(float* p, float v) {
    unsafeAtomicAdd(p, v);   // hardware global_atomic_add_f32 path
}

// ---- A fragment (16x32 bf16) from LDS, row-major bf16 with stride `strideH` halves.
// ISA layout: lane<16 -> M=lane, K = k0..k0+7 (v0..3) and k0+16..k0+23 (v4..7);
//             lane>=16 -> M=lane-16, K = k0+8..k0+15 and k0+24..k0+31.
__device__ __forceinline__ void loadA(const unsigned short* sA, int strideH,
                                      int rowTile, int k0, Frag& a) {
    const int lane = threadIdx.x & 31;
    const int hsel = lane >> 4;
    const int m    = lane & 15;
    const unsigned short* p = sA + (size_t)(rowTile * 16 + m) * strideH + k0 + hsel * 8;
    a.q[0] = *(const uint4*)(p);
    a.q[1] = *(const uint4*)(p + 16);
}

// ---- B fragment (32x16 bf16) from globally pre-packed buffer: per (tileN,kstep,lane)
// 8 contiguous dwords == the fragment registers. Two coalesced b128 loads.
__device__ __forceinline__ void loadB(const uint32_t* Bp, int ksteps,
                                      int tileN, int ks, Frag& b) {
    const int lane = threadIdx.x & 31;
    const uint32_t* p = Bp + (((size_t)(tileN * ksteps + ks) * 32 + lane) << 3);
    b.q[0] = *(const uint4*)(p);
    b.q[1] = *(const uint4*)(p + 4);
}

// 64x128 output tile per block: wave w -> rowTile = w&3, colTiles = (w>>2)*4 .. +3.
template <int KSTEPS>
__device__ __forceinline__ void gemmWave(const unsigned short* sA, int strideH,
                                         const uint32_t* Bp, f32x8 acc[4]) {
    const int wave    = threadIdx.x >> 5;
    const int rowTile = wave & 3;
    const int colBase = (wave >> 2) * 4;
#pragma unroll
    for (int ks = 0; ks < KSTEPS; ++ks) {
        Frag a; loadA(sA, strideH, rowTile, ks * 32, a);
#pragma unroll
        for (int t = 0; t < 4; ++t) {
            Frag b; loadB(Bp, KSTEPS, colBase + t, ks, b);
            acc[t] = __builtin_amdgcn_wmma_f32_16x16x32_bf16(
                false, a.v, false, b.v, (short)0, acc[t], false, false);
        }
    }
}

// ===================== precompute kernels =====================

// W_fused = node_proj_w @ init_proj_w[0:128]
__global__ void kFuseW(const float* __restrict__ npw, const float* __restrict__ ipw,
                       float* __restrict__ Wf) {
    int idx = blockIdx.x * blockDim.x + threadIdx.x;
    if (idx >= HH * HH) return;
    int c = idx >> 7, n = idx & 127;
    float s = 0.f;
    for (int j = 0; j < HH; ++j) s += npw[c * HH + j] * ipw[j * HH + n];
    Wf[c * HH + n] = s;
}

// bias_vec / logp_vec / root_vec folded through init_proj
__global__ void kVecs(const float* __restrict__ npb, const float* __restrict__ lpw,
                      const float* __restrict__ lpb, const float* __restrict__ remb,
                      const float* __restrict__ ipw, const float* __restrict__ ipb,
                      float* biasv, float* lpv, float* rootv) {
    int n = threadIdx.x;
    if (n >= HH) return;
    float b = ipb[n], lv = 0.f, r0 = 0.f, r1 = 0.f;
    for (int j = 0; j < HH; ++j) {
        b  += npb[j] * ipw[j * HH + n] + lpb[j] * ipw[(HH + j) * HH + n];
        lv += lpw[j] * ipw[(HH + j) * HH + n];
        r0 += remb[j]      * ipw[(2 * HH + j) * HH + n];
        r1 += remb[HH + j] * ipw[(2 * HH + j) * HH + n];
    }
    biasv[n] = b; lpv[n] = lv; rootv[n] = r0; rootv[HH + n] = r1;
}

// Pack f32 weight [Kdim x 128] into bf16 WMMA B-fragment order.
__global__ void kPackB(const float* __restrict__ W, int ksteps, uint32_t* __restrict__ out) {
    int total = 8 * ksteps * 32 * 8;
    int idx = blockIdx.x * blockDim.x + threadIdx.x;
    if (idx >= total) return;
    int v = idx & 7;
    int lane = (idx >> 3) & 31;
    int rest = idx >> 8;
    int ks = rest % ksteps;
    int tileN = rest / ksteps;
    int n = tileN * 16 + (lane & 15);
    int k = ks * 32 + ((lane >> 4) * 8) + ((v >> 2) * 16) + 2 * (v & 3);
    uint32_t lo = f2bf(W[(size_t)k * HH + n]);
    uint32_t hi = f2bf(W[(size_t)(k + 1) * HH + n]);
    out[idx] = lo | (hi << 16);
}

__global__ void kClear(float* p, long n) {
    long i = (long)blockIdx.x * blockDim.x + threadIdx.x;
    long stride = (long)gridDim.x * blockDim.x;
    for (; i < n; i += stride) p[i] = 0.f;
}

__global__ void kCount(const int* __restrict__ nodes, float* cnt, int SK) {
    int i = blockIdx.x * blockDim.x + threadIdx.x;
    if (i < SK) { int id = nodes[i]; if (id >= 0) atomAddF(&cnt[id], 1.f); }
}
__global__ void kInv(float* c, int n) {
    int i = blockIdx.x * blockDim.x + threadIdx.x;
    if (i < n) c[i] = 1.f / fmaxf(c[i], 1.f);
}

// ===================== init: h0 = gather(x) @ Wf + lp*lpv + rootv + biasv =====================
__global__ void __launch_bounds__(TPB) kInit(const float* __restrict__ x,
                                             const int* __restrict__ nodes,
                                             const int* __restrict__ rootl,
                                             const float* __restrict__ logp,
                                             const uint32_t* __restrict__ WfP,
                                             const float* __restrict__ biasv,
                                             const float* __restrict__ lpv,
                                             const float* __restrict__ rootv,
                                             float* __restrict__ h) {
    __shared__ __align__(16) unsigned short sA[KK * 136];
    const int s = blockIdx.x, tid = threadIdx.x;
    for (int i = tid; i < KK * HH; i += TPB) {
        int r = i >> 7, c = i & 127;
        int id = nodes[s * KK + r];
        float v = (id >= 0) ? x[(size_t)id * HH + c] : 0.f;
        sA[r * 136 + c] = f2bf(v);
    }
    __syncthreads();
    f32x8 acc[4];
#pragma unroll
    for (int t = 0; t < 4; ++t) acc[t] = zero8();
    gemmWave<4>(sA, 136, WfP, acc);

    const float lp = logp[s];
    const int rl = rootl[s];
    const int lane = tid & 31, wave = tid >> 5;
    const int rowTile = wave & 3, colBase = (wave >> 2) * 4;
#pragma unroll
    for (int t = 0; t < 4; ++t) {
        int col = (colBase + t) * 16 + (lane & 15);
        float add = lp * lpv[col] + biasv[col];
#pragma unroll
        for (int r = 0; r < 8; ++r) {
            int row = rowTile * 16 + r + ((lane >> 4) * 8);
            float rv = rootv[(row == rl ? HH : 0) + col];
            h[((size_t)s * KK + row) * HH + col] = acc[t][r] + add + rv;
        }
    }
}

// ===================== layer part 1: edge-agg + GIN MLP (2 WMMA GEMMs) =====================
__global__ void __launch_bounds__(TPB) kLayer1(const float* __restrict__ h,
                                               const int* __restrict__ nodes,
                                               const int* __restrict__ eidx,
                                               const int* __restrict__ eptr, long E,
                                               const uint32_t* __restrict__ w1p,
                                               const uint32_t* __restrict__ w2p,
                                               const float* __restrict__ b1,
                                               const float* __restrict__ b2,
                                               const float* __restrict__ epsArr, int l,
                                               float* __restrict__ zbuf,
                                               float* __restrict__ xagg) {
    __shared__ __align__(16) unsigned short sA[KK * 136];
    __shared__ __align__(16) float accF[KK * HH];
    __shared__ int2 eL[512];
    const int s = blockIdx.x, tid = threadIdx.x;

    for (int i = tid; i < KK * HH; i += TPB) {
        int r = i >> 7, c = i & 127;
        sA[r * 136 + c] = f2bf(h[((size_t)s * KK + r) * HH + c]);
        accF[i] = 0.f;
    }
    const int e0 = eptr[s], e1 = eptr[s + 1];
    __syncthreads();

    for (int base = e0; base < e1; base += 512) {
        int n = e1 - base; if (n > 512) n = 512;
        for (int i = tid; i < n; i += TPB) {
            eL[i].x = eidx[base + i];          // src local
            eL[i].y = eidx[E + base + i];      // dst local
        }
        __syncthreads();
        for (long t = tid; t < (long)n * HH; t += TPB) {
            int e = (int)(t >> 7), c = (int)(t & 127);
            atomicAdd(&accF[eL[e].y * HH + c], bf2f(sA[eL[e].x * 136 + c]));  // ds_add_f32
        }
        __syncthreads();
    }

    const float ep1 = 1.f + epsArr[l];
    for (int i = tid; i < KK * HH; i += TPB) {
        int r = i >> 7, c = i & 127;
        float z = ep1 * bf2f(sA[r * 136 + c]) + accF[r * HH + c];
        sA[r * 136 + c] = f2bf(z);
    }
    __syncthreads();

    f32x8 acc[4];
#pragma unroll
    for (int t = 0; t < 4; ++t) acc[t] = zero8();
    gemmWave<4>(sA, 136, w1p, acc);

    unsigned short* sY = (unsigned short*)accF;   // reuse accF space (32KB >= 17KB)
    const int lane = tid & 31, wave = tid >> 5;
    const int rowTile = wave & 3, colBase = (wave >> 2) * 4;
#pragma unroll
    for (int t = 0; t < 4; ++t) {
        int col = (colBase + t) * 16 + (lane & 15);
        float bb = b1[l * HH + col];
#pragma unroll
        for (int r = 0; r < 8; ++r) {
            int row = rowTile * 16 + r + ((lane >> 4) * 8);
            float y = acc[t][r] + bb;
            y = y > 0.f ? y : 0.f;
            sY[row * 136 + col] = f2bf(y);
        }
    }
    __syncthreads();

    f32x8 acc2[4];
#pragma unroll
    for (int t = 0; t < 4; ++t) acc2[t] = zero8();
    gemmWave<4>(sY, 136, w2p, acc2);

#pragma unroll
    for (int t = 0; t < 4; ++t) {
        int col = (colBase + t) * 16 + (lane & 15);
        float bb = b2[l * HH + col];
#pragma unroll
        for (int r = 0; r < 8; ++r) {
            int row = rowTile * 16 + r + ((lane >> 4) * 8);
            float z2 = acc2[t][r] + bb;
            zbuf[((size_t)s * KK + row) * HH + col] = z2;
            int id = nodes[s * KK + row];
            if (id >= 0) atomAddF(&xagg[(size_t)id * HH + col], z2);
        }
    }
}

// ===================== layer part 2: mp2 GEMM (K=256) + relu + BN stats =====================
__global__ void __launch_bounds__(TPB) kLayer3(const float* __restrict__ zbuf,
                                               const float* __restrict__ xagg,
                                               const float* __restrict__ invc,
                                               const int* __restrict__ nodes,
                                               const uint32_t* __restrict__ mp2p,
                                               const float* __restrict__ mb, int l,
                                               float* __restrict__ ubuf,
                                               float* gsum, float* gsumsq) {
    __shared__ __align__(16) unsigned short sA[KK * 264];
    const int s = blockIdx.x, tid = threadIdx.x;
    for (int i = tid; i < KK * HH; i += TPB) {
        int r = i >> 7, c = i & 127;
        sA[r * 264 + c] = f2bf(zbuf[((size_t)s * KK + r) * HH + c]);
        int id = nodes[s * KK + r];
        float v = (id >= 0) ? xagg[(size_t)id * HH + c] * invc[id] : 0.f;
        sA[r * 264 + HH + c] = f2bf(v);
    }
    __syncthreads();

    f32x8 acc[4];
#pragma unroll
    for (int t = 0; t < 4; ++t) acc[t] = zero8();
    gemmWave<8>(sA, 264, mp2p, acc);

    const int lane = tid & 31, wave = tid >> 5;
    const int rowTile = wave & 3, colBase = (wave >> 2) * 4;
#pragma unroll
    for (int t = 0; t < 4; ++t) {
        int col = (colBase + t) * 16 + (lane & 15);
        float bb = mb[l * HH + col];
        float ps = 0.f, pss = 0.f;
#pragma unroll
        for (int r = 0; r < 8; ++r) {
            int row = rowTile * 16 + r + ((lane >> 4) * 8);
            float u = acc[t][r] + bb;
            u = u > 0.f ? u : 0.f;
            ubuf[((size_t)s * KK + row) * HH + col] = u;
            ps += u; pss += u * u;
        }
        atomAddF(&gsum[col], ps);
        atomAddF(&gsumsq[col], pss);
    }
}

// ===================== batchnorm + residual =====================
__global__ void kNorm(const float* __restrict__ ubuf, const float* __restrict__ gsum,
                      const float* __restrict__ gsumsq, const float* __restrict__ gamma,
                      const float* __restrict__ beta, int l, float invM,
                      float* __restrict__ h, long total4) {
    long g = (long)blockIdx.x * blockDim.x + threadIdx.x;
    if (g >= total4) return;
    long base = g << 2;
    int c0 = (int)(base & (HH - 1));
    float4 u4 = *(const float4*)(ubuf + base);
    float4 h4 = *(const float4*)(h + base);
    float* up = (float*)&u4;
    float* hp = (float*)&h4;
#pragma unroll
    for (int j = 0; j < 4; ++j) {
        int c = c0 + j;
        float mu = gsum[c] * invM;
        float var = gsumsq[c] * invM - mu * mu;
        float sc = rsqrtf(var + 1e-5f) * gamma[l * HH + c];
        hp[j] = (up[j] - mu) * sc + beta[l * HH + c] + hp[j];
    }
    *(float4*)(h + base) = h4;
}

// ===================== host launch =====================
extern "C" void kernel_launch(void* const* d_in, const int* in_sizes, int n_in,
                              void* d_out, int out_size, void* d_ws, size_t ws_size,
                              hipStream_t stream) {
    (void)n_in; (void)out_size; (void)ws_size;
    const float* x    = (const float*)d_in[0];
    const float* logp = (const float*)d_in[1];
    const float* npw  = (const float*)d_in[2];
    const float* npb  = (const float*)d_in[3];
    const float* lpw  = (const float*)d_in[4];
    const float* lpb  = (const float*)d_in[5];
    const float* remb = (const float*)d_in[6];
    const float* ipw  = (const float*)d_in[7];
    const float* ipb  = (const float*)d_in[8];
    const float* geps = (const float*)d_in[9];
    const float* w1   = (const float*)d_in[10];
    const float* b1   = (const float*)d_in[11];
    const float* w2   = (const float*)d_in[12];
    const float* b2   = (const float*)d_in[13];
    const float* mp2w = (const float*)d_in[14];
    const float* mp2b = (const float*)d_in[15];
    const float* gam  = (const float*)d_in[16];
    const float* bet  = (const float*)d_in[17];
    const int* nodes  = (const int*)d_in[18];
    const int* rootl  = (const int*)d_in[19];
    const int* eidx   = (const int*)d_in[20];
    const int* eptr   = (const int*)d_in[21];

    const int  S    = in_sizes[1];
    const int  Ntot = in_sizes[0] / HH;
    const int  L    = in_sizes[9];
    const long E    = (long)in_sizes[20] / 2;
    const long SK   = (long)S * KK;

    char* ws = (char*)d_ws;
    size_t off = 0;
    auto take = [&](size_t bytes) -> char* {
        char* p = ws + off;
        off = (off + bytes + 255) & ~(size_t)255;
        return p;
    };
    float* zbuf  = (float*)take(SK * HH * sizeof(float));
    float* ubuf  = (float*)take(SK * HH * sizeof(float));
    float* xagg  = (float*)take(((size_t)Ntot * HH + 2 * HH) * sizeof(float));
    float* gsum  = xagg + (size_t)Ntot * HH;
    float* gsumsq = gsum + HH;
    float* invc  = (float*)take((size_t)Ntot * sizeof(float));
    float* WfF   = (float*)take(HH * HH * sizeof(float));
    float* biasv = (float*)take(HH * sizeof(float));
    float* lpv   = (float*)take(HH * sizeof(float));
    float* rootv = (float*)take(2 * HH * sizeof(float));
    uint32_t* WfP = (uint32_t*)take((size_t)HH * HH * 2);
    uint32_t* w1p[8]; uint32_t* w2p[8]; uint32_t* mpp[8];
    for (int l = 0; l < L && l < 8; ++l) {
        w1p[l] = (uint32_t*)take((size_t)HH * HH * 2);
        w2p[l] = (uint32_t*)take((size_t)HH * HH * 2);
        mpp[l] = (uint32_t*)take((size_t)2 * HH * HH * 2);
    }
    float* h = (float*)d_out;

    // --- weight precompute / packing ---
    kFuseW<<<(HH * HH + TPB - 1) / TPB, TPB, 0, stream>>>(npw, ipw, WfF);
    kVecs<<<1, HH, 0, stream>>>(npb, lpw, lpb, remb, ipw, ipb, biasv, lpv, rootv);
    kPackB<<<(8 * 4 * 32 * 8 + TPB - 1) / TPB, TPB, 0, stream>>>(WfF, 4, WfP);
    for (int l = 0; l < L && l < 8; ++l) {
        kPackB<<<(8 * 4 * 32 * 8 + TPB - 1) / TPB, TPB, 0, stream>>>(w1 + (size_t)l * HH * HH, 4, w1p[l]);
        kPackB<<<(8 * 4 * 32 * 8 + TPB - 1) / TPB, TPB, 0, stream>>>(w2 + (size_t)l * HH * HH, 4, w2p[l]);
        kPackB<<<(8 * 8 * 32 * 8 + TPB - 1) / TPB, TPB, 0, stream>>>(mp2w + (size_t)l * 2 * HH * HH, 8, mpp[l]);
    }

    // --- cross-subgraph counts (once) ---
    kClear<<<256, TPB, 0, stream>>>(invc, (long)Ntot);
    kCount<<<(int)((SK + TPB - 1) / TPB), TPB, 0, stream>>>(nodes, invc, (int)SK);
    kInv<<<(Ntot + TPB - 1) / TPB, TPB, 0, stream>>>(invc, Ntot);

    // --- init features (fused gather-GEMM) ---
    kInit<<<S, TPB, 0, stream>>>(x, nodes, rootl, logp, WfP, biasv, lpv, rootv, h);

    const float invM = 1.f / (float)SK;
    const long total4 = SK * HH / 4;
    for (int l = 0; l < L; ++l) {
        int li = l < 8 ? l : 7;
        kClear<<<512, TPB, 0, stream>>>(xagg, (long)Ntot * HH + 2 * HH);
        kLayer1<<<S, TPB, 0, stream>>>(h, nodes, eidx, eptr, E, w1p[li], w2p[li],
                                       b1, b2, geps, l, zbuf, xagg);
        kLayer3<<<S, TPB, 0, stream>>>(zbuf, xagg, invc, nodes, mpp[li], mp2b, l,
                                       ubuf, gsum, gsumsq);
        kNorm<<<(int)((total4 + TPB - 1) / TPB), TPB, 0, stream>>>(
            ubuf, gsum, gsumsq, gam, bet, l, invM, h, total4);
    }
}
